// ConvPointnet_62388694942138
// MI455X (gfx1250) — matrix operand order repaired and checked by reading, not compiled, not run on hardware
//
#include <hip/hip_runtime.h>
#include <hip/hip_bf16.h>

typedef __attribute__((ext_vector_type(16))) _Float16 v16h;
typedef __attribute__((ext_vector_type(8)))  _Float16 v8h;
typedef __attribute__((ext_vector_type(8)))  float    v8f;

#define BB   4
#define TT   32768
#define QQ   8192
#define HH   128
#define CCH  512
#define RRES 64
#define RR2  (RRES * RRES)
#define NN   (BB * TT)
#define PADF 0.1f
#define LDP  136   // padded LDS row stride in f16 (272B = 68 dwords -> bank+4/row, 16B aligned)

// ---------------- WMMA fragment helpers (CDNA5 ISA 7.12.2 layouts) ----------------

// A (16x32 f16): lane L -> row = L%16 ; elements 0..7 -> K = (L/16)*8 + e ;
//                elements 8..15 -> K = 16 + (L/16)*8 + e
template <bool RELU>
__device__ inline v16h a_frag_f32(const float* __restrict__ base, int ld, int lane) {
  const int r = lane & 15, h = lane >> 4;
  const float* p0 = base + (size_t)r * ld + h * 8;
  v16h a;
#pragma unroll
  for (int e = 0; e < 8; ++e) {
    float v0 = p0[e];
    float v1 = p0[16 + e];
    if (RELU) { v0 = fmaxf(v0, 0.0f); v1 = fmaxf(v1, 0.0f); }
    a[e]     = (_Float16)v0;
    a[8 + e] = (_Float16)v1;
  }
  return a;
}

__device__ inline v16h a_frag_f16(const _Float16* base, int ld, int lane) {
  const int r = lane & 15, h = lane >> 4;
  const _Float16* p0 = base + (size_t)r * ld + h * 8;
  v16h a;
#pragma unroll
  for (int e = 0; e < 8; ++e) {
    a[e]     = p0[e];
    a[8 + e] = p0[16 + e];
  }
  return a;
}

// B (32x16 f16) from LDS (padded stride LDP): lane L -> N = n0 + L%16 ; elements 0..7 -> K = k0 + (L/16)*8 + e ;
//                elements 8..15 -> K = k0 + 16 + (L/16)*8 + e
__device__ inline v16h b_frag_lds(const _Float16* Wl, int k0, int n0, int lane) {
  const _Float16* p = Wl + (k0 + (lane >> 4) * 8) * LDP + n0 + (lane & 15);
  v16h b;
#pragma unroll
  for (int e = 0; e < 8; ++e) {
    b[e]     = p[e * LDP];
    b[8 + e] = p[(16 + e) * LDP];
  }
  return b;
}

__device__ inline v8f wmma16(v16h a, v16h b, v8f c) {
  return __builtin_amdgcn_wmma_f32_16x16x32_f16(false, a, false, b, (short)0, c, false, false);
}

// ---------------- small utility kernels ----------------

__global__ void f32_to_f16_kernel(const float* __restrict__ src, _Float16* __restrict__ dst, int n) {
  int i = blockIdx.x * blockDim.x + threadIdx.x;
  if (i < n) dst[i] = (_Float16)src[i];
}

__device__ inline int cell_idx(float a, float b) {
  float u = fminf(fmaxf(a / (1.0f + PADF) + 0.5f, 0.0f), 1.0f - 1e-6f);
  float v = fminf(fmaxf(b / (1.0f + PADF) + 0.5f, 0.0f), 1.0f - 1e-6f);
  int gx = (int)floorf(u * RRES);
  int gy = (int)floorf(v * RRES);
  return gx + RRES * gy;
}

__global__ void index_kernel(const float* __restrict__ P, int* __restrict__ f0,
                             int* __restrict__ f1, int* __restrict__ f2) {
  int t = blockIdx.x * blockDim.x + threadIdx.x;
  if (t >= NN) return;
  float x = P[(size_t)t * 3 + 0];
  float y = P[(size_t)t * 3 + 1];
  float z = P[(size_t)t * 3 + 2];
  int bofs = (t >> 15) * RR2;     // (t / TT) * R2
  f0[t] = bofs + cell_idx(x, z);  // 'xz'
  f1[t] = bofs + cell_idx(x, y);  // 'xy'
  f2[t] = bofs + cell_idx(y, z);  // 'yz'
}

__global__ void fc_pos_kernel(const float* __restrict__ P, const float* __restrict__ W,
                              const float* __restrict__ Bv, float* __restrict__ OUT) {
  int i = blockIdx.x * blockDim.x + threadIdx.x;  // NN*64 threads, 4 cols each
  if (i >= NN * 64) return;
  int t = i >> 6;
  int c0 = (i & 63) * 4;
  float x = P[(size_t)t * 3 + 0];
  float y = P[(size_t)t * 3 + 1];
  float z = P[(size_t)t * 3 + 2];
#pragma unroll
  for (int j = 0; j < 4; ++j) {
    int c = c0 + j;
    OUT[(size_t)t * 256 + c] = x * W[c] + y * W[256 + c] + z * W[512 + c] + Bv[c];
  }
}

// ---------------- ResNet block: out = x@Ws + fc1(relu(fc0(relu(x)))) ----------------
// Weights staged in LDS (padded stride) so B fragments come from ds_load, not L2.

__global__ __launch_bounds__(256) void resnet_block_kernel(
    const float* __restrict__ X,          // (N,256)
    const _Float16* __restrict__ W0, const float* __restrict__ B0,  // 256x128, 128
    const _Float16* __restrict__ W1, const float* __restrict__ B1,  // 128x128, 128
    const _Float16* __restrict__ WS,                                // 256x128
    float* __restrict__ OUT)              // (N,128)
{
  extern __shared__ _Float16 smem[];
  _Float16* sW0 = smem;                      // 256*LDP
  _Float16* sWS = sW0 + 256 * LDP;           // 256*LDP
  _Float16* sW1 = sWS + 256 * LDP;           // 128*LDP
  _Float16* shb0 = sW1 + 128 * LDP;          // 8 waves * 16*LDP

  const int tid  = threadIdx.x;
  const int lane = tid & 31;
  const int w    = tid >> 5;
  const int row0 = (blockIdx.x * 8 + w) * 16;

  // -------- cooperative weight staging (global -> LDS, b128 copies) --------
  {
    const v8h* s0 = (const v8h*)(W0 + (size_t)tid * HH);
    v8h* d0 = (v8h*)(sW0 + tid * LDP);
#pragma unroll
    for (int j = 0; j < 16; ++j) d0[j] = s0[j];
    const v8h* s1 = (const v8h*)(WS + (size_t)tid * HH);
    v8h* d1 = (v8h*)(sWS + tid * LDP);
#pragma unroll
    for (int j = 0; j < 16; ++j) d1[j] = s1[j];
    if (tid < 128) {
      const v8h* s2 = (const v8h*)(W1 + (size_t)tid * HH);
      v8h* d2 = (v8h*)(sW1 + tid * LDP);
#pragma unroll
      for (int j = 0; j < 16; ++j) d2[j] = s2[j];
    }
  }
  __syncthreads();

  const float* Xt = X + (size_t)row0 * 256;
  v8f zero = {};
  v8f acc0[8], accS[8];
#pragma unroll
  for (int n = 0; n < 8; ++n) { acc0[n] = zero; accS[n] = zero; }

  // K = 256 pass: h0 += relu(x) @ W0 ; shortcut += x @ WS
#pragma unroll
  for (int kt = 0; kt < 8; ++kt) {
    v16h arel = a_frag_f32<true >(Xt + kt * 32, 256, lane);
    v16h araw = a_frag_f32<false>(Xt + kt * 32, 256, lane);
#pragma unroll
    for (int n = 0; n < 8; ++n) {
      acc0[n] = wmma16(arel, b_frag_lds(sW0, kt * 32, n * 16, lane), acc0[n]);
      accS[n] = wmma16(araw, b_frag_lds(sWS, kt * 32, n * 16, lane), accS[n]);
    }
  }

  const int col = lane & 15, h = lane >> 4;
  _Float16* hb = shb0 + w * 16 * LDP;
#pragma unroll
  for (int n = 0; n < 8; ++n) {
    float bb = B0[n * 16 + col];
#pragma unroll
    for (int v = 0; v < 8; ++v) {
      float val = fmaxf(acc0[n][v] + bb, 0.0f);           // relu(h0)
      hb[(v + 8 * h) * LDP + n * 16 + col] = (_Float16)val;
    }
  }
  // LDS RAW within a single wave: DS ops are in-order per wave (no barrier needed).

  // K = 128 pass: out += relu(h0) @ W1
#pragma unroll
  for (int kt = 0; kt < 4; ++kt) {
    v16h a = a_frag_f16(hb + kt * 32, LDP, lane);
#pragma unroll
    for (int n = 0; n < 8; ++n)
      accS[n] = wmma16(a, b_frag_lds(sW1, kt * 32, n * 16, lane), accS[n]);
  }

#pragma unroll
  for (int n = 0; n < 8; ++n) {
    float bb = B1[n * 16 + col];
#pragma unroll
    for (int v = 0; v < 8; ++v)
      OUT[(size_t)(row0 + v + 8 * h) * HH + n * 16 + col] = accS[n][v] + bb;
  }
}

// ---------------- scatter max / gather+concat ----------------

__device__ inline unsigned encf(float f) {
  unsigned u = __float_as_uint(f);
  return (u & 0x80000000u) ? ~u : (u | 0x80000000u);
}
__device__ inline float dec0(unsigned u) {
  if (u == 0u) return 0.0f;  // empty cell -> 0 (matches reference)
  unsigned v = (u & 0x80000000u) ? (u & 0x7FFFFFFFu) : ~u;
  return __uint_as_float(v);
}

__global__ void scatter_max_kernel(const float* __restrict__ NET,
                                   const int* __restrict__ f0, const int* __restrict__ f1,
                                   const int* __restrict__ f2,
                                   unsigned* __restrict__ g0, unsigned* __restrict__ g1,
                                   unsigned* __restrict__ g2) {
  int i = blockIdx.x * blockDim.x + threadIdx.x;  // N*128
  if (i >= NN * HH) return;
  int t = i >> 7, c = i & 127;
  unsigned e = encf(NET[i]);
  atomicMax(&g0[(size_t)f0[t] * HH + c], e);
  atomicMax(&g1[(size_t)f1[t] * HH + c], e);
  atomicMax(&g2[(size_t)f2[t] * HH + c], e);
}

__global__ void gather_concat_kernel(const float* __restrict__ NET,
                                     const unsigned* __restrict__ g0, const unsigned* __restrict__ g1,
                                     const unsigned* __restrict__ g2,
                                     const int* __restrict__ f0, const int* __restrict__ f1,
                                     const int* __restrict__ f2,
                                     float* __restrict__ XCAT) {
  int i = blockIdx.x * blockDim.x + threadIdx.x;  // N*128
  if (i >= NN * HH) return;
  int t = i >> 7, c = i & 127;
  float pooled = dec0(g0[(size_t)f0[t] * HH + c]) +
                 dec0(g1[(size_t)f1[t] * HH + c]) +
                 dec0(g2[(size_t)f2[t] * HH + c]);
  XCAT[(size_t)t * 256 + c]       = NET[i];
  XCAT[(size_t)t * 256 + 128 + c] = pooled;
}

// ---------------- fc_c GEMM fused with scatter-add into mean grids ----------------

__global__ __launch_bounds__(256) void fc_c_scatter_kernel(
    const float* __restrict__ X,  // (N,128)
    const _Float16* __restrict__ WC, const float* __restrict__ BC,  // 128x512, 512
    const int* __restrict__ f0, const int* __restrict__ f1, const int* __restrict__ f2,
    float* __restrict__ g0, float* __restrict__ g1, float* __restrict__ g2) {
  __shared__ _Float16 sWC[128 * LDP];   // this block's 128-col slice of WC
  const int tid  = threadIdx.x;
  const int lane = tid & 31;
  const int w    = tid >> 5;
  const int row0 = (blockIdx.x * 8 + w) * 16;
  const int n0   = blockIdx.y * 128;

  {
    int r = tid >> 1, half = (tid & 1) * 64;
    const v8h* s = (const v8h*)(WC + (size_t)r * CCH + n0 + half);
    v8h* d = (v8h*)(sWC + r * LDP + half);
#pragma unroll
    for (int j = 0; j < 8; ++j) d[j] = s[j];
  }
  __syncthreads();

  const float* Xt = X + (size_t)row0 * HH;
  v8f zero = {};
  v8f acc[8];
#pragma unroll
  for (int n = 0; n < 8; ++n) acc[n] = zero;

#pragma unroll
  for (int kt = 0; kt < 4; ++kt) {
    v16h a = a_frag_f32<false>(Xt + kt * 32, HH, lane);
#pragma unroll
    for (int n = 0; n < 8; ++n)
      acc[n] = wmma16(a, b_frag_lds(sWC, kt * 32, n * 16, lane), acc[n]);
  }

  const int col = lane & 15, h = lane >> 4;
#pragma unroll
  for (int n = 0; n < 8; ++n) {
    int cc = n0 + n * 16 + col;
    float bb = BC[cc];
#pragma unroll
    for (int v = 0; v < 8; ++v) {
      int t = row0 + v + 8 * h;
      float val = acc[n][v] + bb;
      atomicAdd(&g0[(size_t)f0[t] * CCH + cc], val);
      atomicAdd(&g1[(size_t)f1[t] * CCH + cc], val);
      atomicAdd(&g2[(size_t)f2[t] * CCH + cc], val);
    }
  }
}

__global__ void count_kernel(const int* __restrict__ f0, const int* __restrict__ f1,
                             const int* __restrict__ f2, float* __restrict__ c0,
                             float* __restrict__ c1, float* __restrict__ c2) {
  int t = blockIdx.x * blockDim.x + threadIdx.x;
  if (t >= NN) return;
  atomicAdd(&c0[f0[t]], 1.0f);
  atomicAdd(&c1[f1[t]], 1.0f);
  atomicAdd(&c2[f2[t]], 1.0f);
}

__global__ void normalize_kernel(float* __restrict__ g, const float* __restrict__ cnt) {
  int i = blockIdx.x * blockDim.x + threadIdx.x;  // B*R2*512
  if (i >= BB * RR2 * CCH) return;
  g[i] /= fmaxf(cnt[i >> 9], 1.0f);
}

// ---------------- bilinear sample (3 planes summed) ----------------

__global__ __launch_bounds__(128) void bilinear_kernel(
    const float* __restrict__ Q3, const float* __restrict__ g0,
    const float* __restrict__ g1, const float* __restrict__ g2,
    float* __restrict__ OUT) {
  int bq = blockIdx.x;  // B*Q blocks
  int b  = bq / QQ;
  float q0 = Q3[(size_t)bq * 3 + 0];
  float q1 = Q3[(size_t)bq * 3 + 1];
  float q2 = Q3[(size_t)bq * 3 + 2];
  const float* gs[3] = {g0, g1, g2};
  const float ua[3]  = {q0, q0, q1};  // xz, xy, yz
  const float va[3]  = {q2, q1, q2};
  int c = threadIdx.x * 4;
  float4 acc = {0.f, 0.f, 0.f, 0.f};
#pragma unroll
  for (int pl = 0; pl < 3; ++pl) {
    float u = fminf(fmaxf(ua[pl] / (1.0f + PADF) + 0.5f, 0.0f), 1.0f - 1e-6f);
    float v = fminf(fmaxf(va[pl] / (1.0f + PADF) + 0.5f, 0.0f), 1.0f - 1e-6f);
    float gx = u * (RRES - 1), gy = v * (RRES - 1);
    float x0f = floorf(gx), y0f = floorf(gy);
    float wx = gx - x0f, wy = gy - y0f;
    int x0 = min(max((int)x0f, 0), RRES - 1), x1 = min(x0 + 1, RRES - 1);
    int y0 = min(max((int)y0f, 0), RRES - 1), y1 = min(y0 + 1, RRES - 1);
    const float* base = gs[pl] + (size_t)b * RR2 * CCH;
    const float4 f00 = *(const float4*)(base + (size_t)(y0 * RRES + x0) * CCH + c);
    const float4 f01 = *(const float4*)(base + (size_t)(y0 * RRES + x1) * CCH + c);
    const float4 f10 = *(const float4*)(base + (size_t)(y1 * RRES + x0) * CCH + c);
    const float4 f11 = *(const float4*)(base + (size_t)(y1 * RRES + x1) * CCH + c);
    float w00 = (1 - wx) * (1 - wy), w01 = wx * (1 - wy);
    float w10 = (1 - wx) * wy,       w11 = wx * wy;
    acc.x += w00 * f00.x + w01 * f01.x + w10 * f10.x + w11 * f11.x;
    acc.y += w00 * f00.y + w01 * f01.y + w10 * f10.y + w11 * f11.y;
    acc.z += w00 * f00.z + w01 * f01.z + w10 * f10.z + w11 * f11.z;
    acc.w += w00 * f00.w + w01 * f01.w + w10 * f10.w + w11 * f11.w;
  }
  *(float4*)(OUT + (size_t)bq * CCH + c) = acc;
}

// ---------------- host orchestration ----------------

extern "C" void kernel_launch(void* const* d_in, const int* in_sizes, int n_in,
                              void* d_out, int out_size, void* d_ws, size_t ws_size,
                              hipStream_t stream) {
  const float* p        = (const float*)d_in[0];
  const float* query    = (const float*)d_in[1];
  const float* fc_pos_w = (const float*)d_in[2];
  const float* fc_pos_b = (const float*)d_in[3];
  const float* blk_w0   = (const float*)d_in[4];
  const float* blk_b0   = (const float*)d_in[5];
  const float* blk_w1   = (const float*)d_in[6];
  const float* blk_b1   = (const float*)d_in[7];
  const float* blk_ws   = (const float*)d_in[8];
  const float* fc_c_w   = (const float*)d_in[9];
  const float* fc_c_b   = (const float*)d_in[10];
  float* out = (float*)d_out;

  char* ws = (char*)d_ws;
  size_t off = 0;
  auto carve = [&](size_t bytes) -> void* {
    off = (off + 255) & ~(size_t)255;
    void* r = ws + off;
    off += bytes;
    return r;
  };

  int*      fidx[3];
  for (int i = 0; i < 3; ++i) fidx[i] = (int*)carve((size_t)NN * 4);
  float*    xcat = (float*)carve((size_t)NN * 256 * 4);
  float*    net  = (float*)carve((size_t)NN * HH * 4);
  unsigned* gmax[3];
  for (int i = 0; i < 3; ++i) gmax[i] = (unsigned*)carve((size_t)BB * RR2 * HH * 4);
  float*    gmean[3];
  for (int i = 0; i < 3; ++i) gmean[i] = (float*)carve((size_t)BB * RR2 * CCH * 4);
  float*    cnt[3];
  for (int i = 0; i < 3; ++i) cnt[i] = (float*)carve((size_t)BB * RR2 * 4);
  _Float16* w0h = (_Float16*)carve((size_t)5 * 256 * HH * 2);
  _Float16* w1h = (_Float16*)carve((size_t)5 * HH * HH * 2);
  _Float16* wsh = (_Float16*)carve((size_t)5 * 256 * HH * 2);
  _Float16* wch = (_Float16*)carve((size_t)HH * CCH * 2);

  // weight conversions f32 -> f16
  {
    int n = 5 * 256 * HH;
    f32_to_f16_kernel<<<(n + 255) / 256, 256, 0, stream>>>(blk_w0, w0h, n);
    f32_to_f16_kernel<<<(n + 255) / 256, 256, 0, stream>>>(blk_ws, wsh, n);
    int n1 = 5 * HH * HH;
    f32_to_f16_kernel<<<(n1 + 255) / 256, 256, 0, stream>>>(blk_w1, w1h, n1);
    int n2 = HH * CCH;
    f32_to_f16_kernel<<<(n2 + 255) / 256, 256, 0, stream>>>(fc_c_w, wch, n2);
  }

  index_kernel<<<(NN + 255) / 256, 256, 0, stream>>>(p, fidx[0], fidx[1], fidx[2]);
  fc_pos_kernel<<<(NN * 64 + 255) / 256, 256, 0, stream>>>(p, fc_pos_w, fc_pos_b, xcat);

  const size_t rb_smem = (size_t)(2 * 256 * LDP + 128 * LDP + 8 * 16 * LDP) * sizeof(_Float16);

  // block 0
  resnet_block_kernel<<<NN / 128, 256, rb_smem, stream>>>(
      xcat, w0h, blk_b0, w1h, blk_b1, wsh, net);

  // blocks 1..4
  for (int i = 1; i < 5; ++i) {
    for (int pl = 0; pl < 3; ++pl)
      hipMemsetAsync(gmax[pl], 0, (size_t)BB * RR2 * HH * 4, stream);
    scatter_max_kernel<<<(NN * HH) / 256, 256, 0, stream>>>(
        net, fidx[0], fidx[1], fidx[2], gmax[0], gmax[1], gmax[2]);
    gather_concat_kernel<<<(NN * HH) / 256, 256, 0, stream>>>(
        net, gmax[0], gmax[1], gmax[2], fidx[0], fidx[1], fidx[2], xcat);
    resnet_block_kernel<<<NN / 128, 256, rb_smem, stream>>>(
        xcat,
        w0h + (size_t)i * 256 * HH, blk_b0 + i * HH,
        w1h + (size_t)i * HH * HH,  blk_b1 + i * HH,
        wsh + (size_t)i * 256 * HH, net);
  }

  // scatter-mean of c = net @ fc_c_w + b, fused
  for (int pl = 0; pl < 3; ++pl) {
    hipMemsetAsync(gmean[pl], 0, (size_t)BB * RR2 * CCH * 4, stream);
    hipMemsetAsync(cnt[pl], 0, (size_t)BB * RR2 * 4, stream);
  }
  count_kernel<<<(NN + 255) / 256, 256, 0, stream>>>(
      fidx[0], fidx[1], fidx[2], cnt[0], cnt[1], cnt[2]);
  {
    dim3 grid(NN / 128, 4);
    fc_c_scatter_kernel<<<grid, 256, 0, stream>>>(
        net, wch, fc_c_b, fidx[0], fidx[1], fidx[2], gmean[0], gmean[1], gmean[2]);
  }
  for (int pl = 0; pl < 3; ++pl)
    normalize_kernel<<<(BB * RR2 * CCH) / 256, 256, 0, stream>>>(gmean[pl], cnt[pl]);

  bilinear_kernel<<<BB * QQ, 128, 0, stream>>>(query, gmean[0], gmean[1], gmean[2], out);

  (void)in_sizes; (void)n_in; (void)out_size; (void)ws_size;
}
